// DomainAwareGAT_35150012351252
// MI455X (gfx1250) — matrix operand order repaired and checked by reading, not compiled, not run on hardware
//
#include <hip/hip_runtime.h>
#include <hip/hip_bf16.h>

typedef __attribute__((ext_vector_type(16))) __bf16 v16bf;
typedef __attribute__((ext_vector_type(8)))  __bf16 v8bf;
typedef __attribute__((ext_vector_type(8)))  float  v8f;

#define GN 20000
#define GE 320000
#define GD 256
#define GH 8

// ---------------- conversion kernels ----------------
__global__ __launch_bounds__(256) void cvt_x_bf16(const float* __restrict__ in,
                                                  __bf16* __restrict__ out, int n) {
    int i = blockIdx.x * 256 + threadIdx.x;
    if (i < n) out[i] = (__bf16)in[i];
}

// W [K=256][J=256] row-major -> WT bf16 [J][K] (so B-fragment loads are contiguous in K)
__global__ __launch_bounds__(256) void cvt_wT_bf16(const float* __restrict__ W,
                                                   __bf16* __restrict__ WT) {
    int i = blockIdx.x * 256 + threadIdx.x;   // i = k*256 + j
    int k = i >> 8, j = i & 255;
    WT[j * 256 + k] = (__bf16)W[k * 256 + j];
}

// ---------------- WMMA GEMM: Y[N,256] = X[N,256] @ W + bias ----------------
// One wave computes one 16x16 tile; 8 x v_wmma_f32_16x16x32_bf16 over K=256.
__global__ __launch_bounds__(256) void gemm_bf16_bias(const __bf16* __restrict__ Xbf,
                                                      const __bf16* __restrict__ WT,
                                                      const float* __restrict__ bias,
                                                      float* __restrict__ Y) {
    const int lane    = threadIdx.x & 31;
    const int wave    = threadIdx.x >> 5;
    const int rowTile = blockIdx.x;                 // 0..1249
    const int colTile = blockIdx.y * 8 + wave;      // 0..15
    const int row = rowTile * 16 + (lane & 15);
    const int col = colTile * 16 + (lane & 15);
    const int kbA = (lane >> 4) * 16;               // A: contiguous K per half-wave
    const int kbB = (lane >> 4) * 16;               // B: contiguous K per half-wave

    v8f c = {};
    const __bf16* arow = Xbf + (size_t)row * 256;
    const __bf16* bcol = WT  + (size_t)col * 256;
    #pragma unroll
    for (int kk = 0; kk < 256; kk += 32) {
        v8bf alo = *(const v8bf*)(arow + kk + kbA);
        v8bf ahi = *(const v8bf*)(arow + kk + kbA + 8);
        v16bf a;
        #pragma unroll
        for (int i = 0; i < 8; ++i) { a[i] = alo[i]; a[i + 8] = ahi[i]; }
        v16bf b = *(const v16bf*)(bcol + kk + kbB);
        c = __builtin_amdgcn_wmma_f32_16x16x32_bf16(false, a, false, b,
                                                    (short)0, c, false, false);
    }
    const int rbase = rowTile * 16 + ((lane >> 4) << 3);  // C/D: M=i (lo half) / M=i+8 (hi half)
    const float bv = bias[col];
    #pragma unroll
    for (int i = 0; i < 8; ++i)
        Y[(size_t)(rbase + i) * 256 + col] = c[i] + bv;
}

// ---------------- edge scoring: logits[e,h] = a_h . leakyrelu(xl[src]+xr[dst]+ea) ----
__global__ __launch_bounds__(256) void edge_logits_k(const float* __restrict__ xl,
                                                     const float* __restrict__ xr,
                                                     const int* __restrict__ src,
                                                     const int* __restrict__ dst,
                                                     const float* __restrict__ eattr,
                                                     const float* __restrict__ We,
                                                     const float* __restrict__ att,
                                                     float* __restrict__ logits) {
    const int wave = threadIdx.x >> 5, lane = threadIdx.x & 31;
    const int e = blockIdx.x * 8 + wave;
    const int s = src[e], d = dst[e];
    const float ea = eattr[e];
    const float* pl = xl + (size_t)s * 256;
    const float* pr = xr + (size_t)d * 256;
    #pragma unroll
    for (int h = 0; h < GH; ++h) {
        const int idx = h * 32 + lane;
        float m = pl[idx] + pr[idx] + ea * We[idx];
        m = m > 0.f ? m : 0.2f * m;                  // LeakyReLU(0.2)
        float t = m * att[idx];
        #pragma unroll
        for (int off = 16; off; off >>= 1) t += __shfl_xor(t, off, 32);
        if (lane == 0) logits[(size_t)e * GH + h] = t;
    }
}

// monotonic float<->uint mapping for atomic max
__device__ __forceinline__ unsigned fmap(float f) {
    unsigned u = __float_as_uint(f);
    return (u & 0x80000000u) ? ~u : (u | 0x80000000u);
}
__device__ __forceinline__ float funmap(unsigned u) {
    unsigned v = (u & 0x80000000u) ? (u & 0x7FFFFFFFu) : ~u;
    return __uint_as_float(v);
}

__global__ __launch_bounds__(256) void init_nodes_k(unsigned* __restrict__ nmax,
                                                    float* __restrict__ den) {
    int i = blockIdx.x * 256 + threadIdx.x;
    nmax[i] = 0u;
    den[i]  = 0.f;
}

__global__ __launch_bounds__(256) void init_agg_k(float* __restrict__ agg,
                                                  const float* __restrict__ bias_out) {
    int i = blockIdx.x * 256 + threadIdx.x;
    agg[i] = bias_out[i & 255];
}

__global__ __launch_bounds__(256) void scatter_max_k(const float* __restrict__ logits,
                                                     const int* __restrict__ dst,
                                                     unsigned* __restrict__ nmax) {
    int i = blockIdx.x * 256 + threadIdx.x;     // i = e*8 + h
    int e = i >> 3, h = i & 7;
    atomicMax(&nmax[(size_t)dst[e] * GH + h], fmap(logits[i]));
}

__global__ __launch_bounds__(256) void exp_den_k(float* __restrict__ logits,
                                                 const int* __restrict__ dst,
                                                 const unsigned* __restrict__ nmax,
                                                 float* __restrict__ den) {
    int i = blockIdx.x * 256 + threadIdx.x;
    int e = i >> 3, h = i & 7;
    int d = dst[e];
    float ex = __expf(logits[i] - funmap(nmax[(size_t)d * GH + h]));
    atomicAdd(&den[(size_t)d * GH + h], ex);
    logits[i] = ex;                              // overwrite with numerator
}

__global__ __launch_bounds__(256) void aggregate_k(const float* __restrict__ xl,
                                                   const int* __restrict__ src,
                                                   const int* __restrict__ dst,
                                                   const float* __restrict__ exbuf,
                                                   const float* __restrict__ den,
                                                   float* __restrict__ agg) {
    const int wave = threadIdx.x >> 5, lane = threadIdx.x & 31;
    const int e = blockIdx.x * 8 + wave;
    const int s = src[e], d = dst[e];
    const float* pl = xl + (size_t)s * 256;
    float* pa = agg + (size_t)d * 256;
    #pragma unroll
    for (int h = 0; h < GH; ++h) {
        float alpha = exbuf[(size_t)e * GH + h] / (den[(size_t)d * GH + h] + 1e-16f);
        atomicAdd(&pa[h * 32 + lane], pl[h * 32 + lane] * alpha);
    }
}

// ---------------- LayerNorm + ELU + residual ----------------
__global__ __launch_bounds__(256) void post_ln_k(const float* __restrict__ agg,
                                                 const float* __restrict__ xin,
                                                 const float* __restrict__ g,
                                                 const float* __restrict__ b,
                                                 float* __restrict__ xout) {
    __shared__ float red[8];
    const int n = blockIdx.x, d = threadIdx.x;
    const int lane = d & 31, w = d >> 5;
    const float v = agg[(size_t)n * 256 + d];
    float t = v;
    #pragma unroll
    for (int off = 16; off; off >>= 1) t += __shfl_xor(t, off, 32);
    if (lane == 0) red[w] = t;
    __syncthreads();
    float mu = 0.f;
    #pragma unroll
    for (int i = 0; i < 8; ++i) mu += red[i];
    mu *= (1.f / 256.f);
    __syncthreads();
    const float dv = v - mu;
    t = dv * dv;
    #pragma unroll
    for (int off = 16; off; off >>= 1) t += __shfl_xor(t, off, 32);
    if (lane == 0) red[w] = t;
    __syncthreads();
    float var = 0.f;
    #pragma unroll
    for (int i = 0; i < 8; ++i) var += red[i];
    var *= (1.f / 256.f);
    float y = dv * rsqrtf(var + 1e-5f) * g[d] + b[d];
    float e = y > 0.f ? y : (__expf(y) - 1.f);     // ELU
    xout[(size_t)n * 256 + d] = xin[(size_t)n * 256 + d] + e;
}

// ---------------- host launch ----------------
extern "C" void kernel_launch(void* const* d_in, const int* in_sizes, int n_in,
                              void* d_out, int out_size, void* d_ws, size_t ws_size,
                              hipStream_t stream) {
    const float* x        = (const float*)d_in[0];
    const int*   ei       = (const int*)d_in[1];
    const float* eattr    = (const float*)d_in[2];
    const float* Wl       = (const float*)d_in[3];
    const float* bl       = (const float*)d_in[4];
    const float* Wr       = (const float*)d_in[5];
    const float* br       = (const float*)d_in[6];
    const float* We       = (const float*)d_in[7];
    const float* att      = (const float*)d_in[8];
    const float* bias_out = (const float*)d_in[9];
    const float* ln_g     = (const float*)d_in[10];
    const float* ln_b     = (const float*)d_in[11];
    float* out = (float*)d_out;

    const int* src = ei;
    const int* dst = ei + GE;

    char* ws = (char*)d_ws;
    size_t off = 0;
    auto carve = [&](size_t bytes) -> void* {
        void* p = ws + off;
        off = (off + bytes + 255) & ~(size_t)255;
        return p;
    };
    __bf16*  xbf    = (__bf16*)carve((size_t)GN * GD * 2);
    __bf16*  wlT    = (__bf16*)carve((size_t)GD * GD * 2);
    __bf16*  wrT    = (__bf16*)carve((size_t)GD * GD * 2);
    float*   xl     = (float*)carve((size_t)GN * GD * 4);
    float*   xr     = (float*)carve((size_t)GN * GD * 4);
    float*   agg    = (float*)carve((size_t)GN * GD * 4);
    float*   logits = (float*)carve((size_t)GE * GH * 4);
    unsigned* nmax  = (unsigned*)carve((size_t)GN * GH * 4);
    float*   den    = (float*)carve((size_t)GN * GH * 4);

    for (int i = 0; i < 2; ++i) {
        const float* xin  = (i == 0) ? x : out;
        float*       xout = out;                      // layer0 -> out, layer1 in-place

        cvt_x_bf16<<<(GN * GD) / 256, 256, 0, stream>>>(xin, xbf, GN * GD);
        cvt_wT_bf16<<<(GD * GD) / 256, 256, 0, stream>>>(Wl + (size_t)i * GD * GD, wlT);
        cvt_wT_bf16<<<(GD * GD) / 256, 256, 0, stream>>>(Wr + (size_t)i * GD * GD, wrT);

        gemm_bf16_bias<<<dim3(GN / 16, 2), 256, 0, stream>>>(xbf, wlT, bl + i * GD, xl);
        gemm_bf16_bias<<<dim3(GN / 16, 2), 256, 0, stream>>>(xbf, wrT, br + i * GD, xr);

        edge_logits_k<<<GE / 8, 256, 0, stream>>>(xl, xr, src, dst, eattr,
                                                  We + i * GD, att + i * GD, logits);

        init_nodes_k<<<(GN * GH) / 256, 256, 0, stream>>>(nmax, den);
        init_agg_k<<<(GN * GD) / 256, 256, 0, stream>>>(agg, bias_out + i * GD);

        scatter_max_k<<<(GE * GH) / 256, 256, 0, stream>>>(logits, dst, nmax);
        exp_den_k<<<(GE * GH) / 256, 256, 0, stream>>>(logits, dst, nmax, den);
        aggregate_k<<<GE / 8, 256, 0, stream>>>(xl, src, dst, logits, den, agg);

        post_ln_k<<<GN, 256, 0, stream>>>(agg, xin, ln_g + i * GD, ln_b + i * GD, xout);
    }
}
